// TwoLayerGCN_61143154426381
// MI455X (gfx1250) — matrix operand order repaired and checked by reading, not compiled, not run on hardware
//
#include <hip/hip_runtime.h>

// ---------------------------------------------------------------------------
// Two-layer GCN for MI455X (gfx1250, wave32, WMMA, double-buffered async-to-LDS).
//  pass 0: x -> bf16 [M][K], W1 -> bf16 [N][K] (transposed), W2 -> bf16T
//  pass 1: h1 = x_bf @ W1T + b1          (v_wmma_f32_16x16x32_bf16)
//  pass 2: ag1 = A_hat spmm h1           (L2 gather + global_atomic_add_f32)
//  pass 3: ag_bf = relu(ag1) -> bf16
//  pass 4: h2 = ag_bf @ W2T + b2
//  pass 5: out = A_hat spmm h2
// GEMM tile fill: GLOBAL_LOAD_ASYNC_TO_LDS_B128 on ASYNCcnt, software-
// pipelined across two LDS buffers (wait ASYNCcnt<=PER keeps the next tile's
// copies in flight behind the WMMA work).
// ---------------------------------------------------------------------------

typedef __attribute__((ext_vector_type(16))) __bf16 v16bf;
typedef __attribute__((ext_vector_type(8)))  __bf16 v8bf;
typedef __attribute__((ext_vector_type(4)))  __bf16 v4bf;
typedef __attribute__((ext_vector_type(8)))  float  v8f;

static __device__ __forceinline__ __bf16 f2bf(float f) {
    // round-to-nearest-even truncation to bf16 (bandwidth-bound passes only)
    union { float f; unsigned u; } x; x.f = f;
    unsigned r = (x.u + 0x7fffu + ((x.u >> 16) & 1u)) >> 16;
    unsigned short s = (unsigned short)r;
    return __builtin_bit_cast(__bf16, s);
}

static __device__ __forceinline__ v16bf cat16(v8bf a, v8bf b) {
    return __builtin_shufflevector(a, b, 0,1,2,3,4,5,6,7,8,9,10,11,12,13,14,15);
}

// CDNA5 async copy: 16B global -> LDS per lane, tracked on ASYNCcnt.
static __device__ __forceinline__ void async_b128(const __bf16* gptr, const __bf16* lptr) {
    unsigned lds = (unsigned)(uintptr_t)lptr;   // generic LDS ptr low 32 bits = LDS offset
    asm volatile("global_load_async_to_lds_b128 %0, %1, off"
                 :: "v"(lds), "v"(gptr) : "memory");
}
template<int N>
static __device__ __forceinline__ void wait_async_le() {
    asm volatile("s_wait_asynccnt %0" :: "n"(N) : "memory");
}

// ---------------------------------------------------------------------------
// GEMM: out[M,N] = A_bf16[M,K] @ WT_bf16[N,K]^T + bias.
// 128 threads = 4 waves (2x2); block tile 64 x BN; BK=64; wave tile 32 x BN/2.
// Double-buffered LDS; per-wave PER async copies per tile.
// ---------------------------------------------------------------------------
template<int BN>
__global__ __launch_bounds__(128)
void gcn_gemm_bf16(const __bf16* __restrict__ A,   // [M][K]
                   const __bf16* __restrict__ WT,  // [N][K]
                   const float*  __restrict__ bias,
                   float*        __restrict__ out,
                   int M, int K, int N)
{
    constexpr int BK  = 64;
    constexpr int KST = 72;              // LDS row stride in bf16 (144B, 16B-aligned spans)
    constexpr int NWT = BN / 32;         // wmma n-tiles per wave (4 for BN=128, 2 for BN=64)
    constexpr int ACH = (64 * BK) / (8 * 128);   // 16B chunks per thread, A tile (=4)
    constexpr int BCH = (BN * BK) / (8 * 128);   // 16B chunks per thread, B tile (8/4)
    constexpr int PER = ACH + BCH;               // per-thread async copies per tile

    __shared__ __bf16 Alds[2][64][KST] __attribute__((aligned(16)));   // [buf][m][k]
    __shared__ __bf16 Blds[2][BN][KST] __attribute__((aligned(16)));   // [buf][n][k]

    const int tid  = threadIdx.x;
    const int lane = tid & 31;
    const int wave = tid >> 5;
    const int l    = lane & 15;
    const int hi   = lane >> 4;
    const int wy   = wave >> 1;          // wave row (M)
    const int wx   = wave & 1;           // wave col (N)

    const int mBase = blockIdx.x * 64;
    const int nBase = blockIdx.y * BN;

    v8f acc[2][NWT] = {};

    // issue all async copies for the tile at k-offset kk into buffer b
    auto issueTile = [&](int kk, int b) {
        #pragma unroll
        for (int i = 0; i < ACH; ++i) {
            int c  = tid + i * 128;
            int r  = c >> 3;             // row 0..63 (8 chunks per 64-elem row)
            int cc = (c & 7) * 8;        // bf16 col
            int gr = min(mBase + r, M - 1);   // clamp: dup row for OOB, store-guarded later
            async_b128(A + (size_t)gr * K + kk + cc, &Alds[b][r][cc]);
        }
        #pragma unroll
        for (int i = 0; i < BCH; ++i) {
            int c  = tid + i * 128;
            int r  = c >> 3;             // row n 0..BN-1
            int cc = (c & 7) * 8;
            async_b128(WT + (size_t)(nBase + r) * K + kk + cc, &Blds[b][r][cc]);
        }
    };

    issueTile(0, 0);                      // prologue: tile 0 in flight

    for (int kk = 0; kk < K; kk += BK) {
        const int cur = (kk / BK) & 1;
        // issue next tile into the other buffer (its last readers finished at
        // the barrier that ended the previous iteration). Last iteration
        // re-issues an in-bounds redundant tile to stay branchless.
        const int knext = (kk + BK < K) ? (kk + BK) : kk;
        issueTile(knext, cur ^ 1);

        wait_async_le<PER>();             // current tile's copies have landed;
        __syncthreads();                  // next tile's stay in flight

        #pragma unroll
        for (int ks = 0; ks < BK; ks += 32) {
            // A 16x32 frag (ISA 7.12.2): lanes 0-15 K 0-7/16-23, lanes 16-31 K 8-15/24-31
            v16bf af[2];
            #pragma unroll
            for (int mt = 0; mt < 2; ++mt) {
                int m = wy * 32 + mt * 16 + l;
                v8bf p0 = *reinterpret_cast<const v8bf*>(&Alds[cur][m][ks + hi * 8]);
                v8bf p1 = *reinterpret_cast<const v8bf*>(&Alds[cur][m][ks + 16 + hi * 8]);
                af[mt] = cat16(p0, p1);
            }
            // B 32x16 frag: lane = column n; lanes 0-15 K 0-7/16-23, lanes 16-31 K 8-15/24-31
            v16bf bfg[NWT];
            #pragma unroll
            for (int nt = 0; nt < NWT; ++nt) {
                int n = wx * (BN / 2) + nt * 16 + l;
                v8bf p0 = *reinterpret_cast<const v8bf*>(&Blds[cur][n][ks + hi * 8]);
                v8bf p1 = *reinterpret_cast<const v8bf*>(&Blds[cur][n][ks + 16 + hi * 8]);
                bfg[nt] = cat16(p0, p1);
            }
            #pragma unroll
            for (int mt = 0; mt < 2; ++mt)
                #pragma unroll
                for (int nt = 0; nt < NWT; ++nt)
                    acc[mt][nt] = __builtin_amdgcn_wmma_f32_16x16x32_bf16(
                        false, af[mt], false, bfg[nt],
                        (short)0, acc[mt][nt], false, false);
        }
        __syncthreads();                  // all waves done reading buf[cur]
    }

    // --- epilogue: C/D layout -> VGPR r holds row hi*8+r, col = lane&15 ---
    #pragma unroll
    for (int mt = 0; mt < 2; ++mt) {
        #pragma unroll
        for (int nt = 0; nt < NWT; ++nt) {
            int col = nBase + wx * (BN / 2) + nt * 16 + l;
            float b = bias[col];
            #pragma unroll
            for (int r8 = 0; r8 < 8; ++r8) {
                int row = mBase + wy * 32 + mt * 16 + hi * 8 + r8;
                if (row < M) out[(size_t)row * N + col] = acc[mt][nt][r8] + b;
            }
        }
    }
}

// ---------------------------------------------------------------------------
// Conversion passes (bandwidth-bound; f2bf VALU cost hidden under HBM/L2).
// ---------------------------------------------------------------------------
__global__ __launch_bounds__(256)
void gcn_cvt_bf16(const float4* __restrict__ in, v4bf* __restrict__ out,
                  int n4, int relu)
{
    int i = blockIdx.x * 256 + threadIdx.x;
    if (i >= n4) return;
    float4 v = in[i];
    if (relu) {
        v.x = fmaxf(v.x, 0.f); v.y = fmaxf(v.y, 0.f);
        v.z = fmaxf(v.z, 0.f); v.w = fmaxf(v.w, 0.f);
    }
    v4bf o;
    o[0] = f2bf(v.x); o[1] = f2bf(v.y); o[2] = f2bf(v.z); o[3] = f2bf(v.w);
    out[i] = o;
}

// W[K][N] f32 -> WT[N][K] bf16 (tiny matrices; output-coalesced)
__global__ __launch_bounds__(256)
void gcn_cvt_tr(const float* __restrict__ W, __bf16* __restrict__ WT, int K, int N)
{
    int idx = blockIdx.x * 256 + threadIdx.x;
    if (idx >= K * N) return;
    int n = idx / K;
    int k = idx - n * K;
    WT[idx] = f2bf(W[(size_t)k * N + n]);
}

// ---------------------------------------------------------------------------
// SpMM scatter: gathers from L2-resident h (global_load_b128), scatters with
// fire-and-forget global_atomic_add_f32 (no-return -> STOREcnt only).
// ---------------------------------------------------------------------------
__global__ __launch_bounds__(256)
void gcn_spmm_d256(const int* __restrict__ rows, const int* __restrict__ cols,
                   const float* __restrict__ vals, const float* __restrict__ h,
                   float* __restrict__ out, int E)
{
    int wid  = (blockIdx.x * 256 + threadIdx.x) >> 5;
    int lane = threadIdx.x & 31;
    if (wid >= E) return;
    int   r = rows[wid];
    int   c = cols[wid];
    float v = vals[wid];
    const float* hr = h   + (size_t)c * 256;
    float*       o  = out + (size_t)r * 256;
    #pragma unroll
    for (int d = lane * 4; d < 256; d += 128) {
        float4 hv = *reinterpret_cast<const float4*>(hr + d);
        atomicAdd(o + d + 0, v * hv.x);
        atomicAdd(o + d + 1, v * hv.y);
        atomicAdd(o + d + 2, v * hv.z);
        atomicAdd(o + d + 3, v * hv.w);
    }
}

__global__ __launch_bounds__(256)
void gcn_spmm_d64(const int* __restrict__ rows, const int* __restrict__ cols,
                  const float* __restrict__ vals, const float* __restrict__ h,
                  float* __restrict__ out, int E)
{
    int wid  = (blockIdx.x * 256 + threadIdx.x) >> 5;
    int lane = threadIdx.x & 31;
    int e    = wid * 2 + (lane >> 4);     // two edges per wave
    int l16  = lane & 15;
    if (e >= E) return;
    int   r = rows[e];
    int   c = cols[e];
    float v = vals[e];
    float4 hv = *reinterpret_cast<const float4*>(h + (size_t)c * 64 + l16 * 4);
    float* o  = out + (size_t)r * 64 + l16 * 4;
    atomicAdd(o + 0, v * hv.x);
    atomicAdd(o + 1, v * hv.y);
    atomicAdd(o + 2, v * hv.z);
    atomicAdd(o + 3, v * hv.w);
}

__global__ __launch_bounds__(256)
void gcn_zero_f32(float4* __restrict__ p, int n4)
{
    int i = blockIdx.x * 256 + threadIdx.x;
    if (i < n4) p[i] = make_float4(0.f, 0.f, 0.f, 0.f);
}

// ---------------------------------------------------------------------------
extern "C" void kernel_launch(void* const* d_in, const int* in_sizes, int n_in,
                              void* d_out, int out_size, void* d_ws, size_t ws_size,
                              hipStream_t stream)
{
    const float* x   = (const float*)d_in[0];
    const int*   er  = (const int*)  d_in[1];
    const int*   ec  = (const int*)  d_in[2];
    const float* ev  = (const float*)d_in[3];
    const float* W1  = (const float*)d_in[4];
    const float* b1  = (const float*)d_in[5];
    const float* W2  = (const float*)d_in[6];
    const float* b2  = (const float*)d_in[7];
    float*       outp = (float*)d_out;

    const int M    = 50000;
    const int DIN  = 512;
    const int DHID = 256;
    const int DOUT = 64;
    const int E    = in_sizes[1];

    // ---- workspace layout (aliasing keeps peak ~154 MB) ----
    char* wsp = (char*)d_ws;
    float* h1f = (float*)wsp;                 wsp += (size_t)M * DHID * 4;  // 51.2 MB
    float* ag1 = (float*)wsp;                 wsp += (size_t)M * DHID * 4;  // 51.2 MB
    __bf16* xbf = (__bf16*)wsp;               wsp += (size_t)M * DIN * 2;   // 51.2 MB
    __bf16* w1t = (__bf16*)wsp;               wsp += (size_t)DIN * DHID * 2;
    __bf16* w2t = (__bf16*)wsp;               wsp += (size_t)DHID * DOUT * 2;
    float*  h2f  = h1f;                       // h1 dead after spmm1
    __bf16* agbf = xbf;                       // x_bf16 dead after gemm1

    // zero scatter targets
    {
        int n4 = (M * DHID) / 4;
        gcn_zero_f32<<<(n4 + 255) / 256, 256, 0, stream>>>((float4*)ag1, n4);
        int m4 = (M * DOUT) / 4;
        gcn_zero_f32<<<(m4 + 255) / 256, 256, 0, stream>>>((float4*)outp, m4);
    }
    // pass 0: conversions
    {
        int n4 = (M * DIN) / 4;
        gcn_cvt_bf16<<<(n4 + 255) / 256, 256, 0, stream>>>((const float4*)x, (v4bf*)xbf, n4, 0);
        int w1n = DIN * DHID;
        gcn_cvt_tr<<<(w1n + 255) / 256, 256, 0, stream>>>(W1, w1t, DIN, DHID);
        int w2n = DHID * DOUT;
        gcn_cvt_tr<<<(w2n + 255) / 256, 256, 0, stream>>>(W2, w2t, DHID, DOUT);
    }
    // pass 1: h1 = x @ W1 + b1
    {
        dim3 grid((M + 63) / 64, DHID / 128);
        gcn_gemm_bf16<128><<<grid, 128, 0, stream>>>(xbf, w1t, b1, h1f, M, DIN, DHID);
    }
    // pass 2: ag1 = A_hat @ h1
    {
        int blocks = (E * 32 + 255) / 256;    // one wave per edge
        gcn_spmm_d256<<<blocks, 256, 0, stream>>>(er, ec, ev, h1f, ag1, E);
    }
    // pass 3: ag_bf = relu(ag1) as bf16
    {
        int n4 = (M * DHID) / 4;
        gcn_cvt_bf16<<<(n4 + 255) / 256, 256, 0, stream>>>((const float4*)ag1, (v4bf*)agbf, n4, 1);
    }
    // pass 4: h2 = ag_bf @ W2 + b2
    {
        dim3 grid((M + 63) / 64, DOUT / 64);
        gcn_gemm_bf16<64><<<grid, 128, 0, stream>>>(agbf, w2t, b2, h2f, M, DHID, DOUT);
    }
    // pass 5: out = A_hat @ h2
    {
        int waves  = (E + 1) / 2;             // two edges per wave
        int blocks = (waves * 32 + 255) / 256;
        gcn_spmm_d64<<<blocks, 256, 0, stream>>>(er, ec, ev, h2f, outp, E);
    }
}
